// NovaMindMoELayer_16887811408649
// MI455X (gfx1250) — matrix operand order repaired and checked by reading, not compiled, not run on hardware
//
#include <hip/hip_runtime.h>

typedef __bf16 bf16;
typedef __bf16 v8bf  __attribute__((ext_vector_type(8)));
typedef __bf16 v16bf __attribute__((ext_vector_type(16)));
typedef float  v8f   __attribute__((ext_vector_type(8)));

constexpr int N_TOK = 8192;   // B*T
constexpr int DIM   = 1024;   // D
constexpr int NEXP  = 16;     // E
constexpr int IEXP  = 512;    // I
constexpr int ISH   = 2048;   // IS
constexpr int TOPK  = 2;      // K
constexpr float ALPHA_C = 1e-4f;
constexpr float EPS_C   = 1e-9f;

constexpr int BM = 128, BN = 64, BK = 64;
constexpr int LDT = 72;                 // LDS row stride (bf16): 144B rows, 16B-aligned, A-frags bank-conflict-free
constexpr int AITER = (BM * BK / 8) / 256;  // 4 x 16B chunks per thread for A tile
constexpr int BITER = (BN * BK / 8) / 256;  // 2 x 16B chunks per thread per B tile

// ---------------- helpers ----------------
static __device__ __forceinline__ v16bf ld_frag(const bf16* p0, const bf16* p1) {
  v8bf lo = *(const v8bf*)p0;
  v8bf hi = *(const v8bf*)p1;
  return __builtin_shufflevector(lo, hi, 0,1,2,3,4,5,6,7,8,9,10,11,12,13,14,15);
}

// ---------------- elementwise f32 -> bf16 ----------------
__global__ void cvt_f32_bf16(const float* __restrict__ in, bf16* __restrict__ out, long n) {
  long i = (long)blockIdx.x * blockDim.x + threadIdx.x;
  if (i < n) out[i] = (bf16)in[i];
}

// ---------------- tiled transpose + convert: in[z][R][C] f32 -> out[z][C][R] bf16 ----------------
__global__ void transpose_cvt(const float* __restrict__ in, bf16* __restrict__ out, int R, int C) {
  __shared__ float tile[32][33];
  long z = blockIdx.z;
  const float* inz = in + z * (long)R * C;
  bf16* outz = out + z * (long)R * C;
  int c0 = blockIdx.x * 32, r0 = blockIdx.y * 32;
#pragma unroll
  for (int i = 0; i < 4; ++i)
    tile[threadIdx.y + i * 8][threadIdx.x] =
        inz[(long)(r0 + threadIdx.y + i * 8) * C + c0 + threadIdx.x];
  __syncthreads();
#pragma unroll
  for (int i = 0; i < 4; ++i)
    outz[(long)(c0 + threadIdx.y + i * 8) * R + r0 + threadIdx.x] =
        (bf16)tile[threadIdx.x][threadIdx.y + i * 8];
}

// ---------------- router: affinity = sigmoid(x @ gate_w), one thread per (n,e) ----------------
__global__ void router_aff(const float* __restrict__ x, const float* __restrict__ gw,
                           float* __restrict__ aff) {
  int id = blockIdx.x * 256 + threadIdx.x;
  if (id >= N_TOK * NEXP) return;
  int n = id >> 4, e = id & 15;
  const float* xr = x + (long)n * DIM;
  float s = 0.f;
  for (int d = 0; d < DIM; ++d) s += xr[d] * gw[d * NEXP + e];
  aff[id] = 1.f / (1.f + __expf(-s));
}

// ---------------- top-2 routing, gate normalization, per-expert lists, P-sum ----------------
__global__ void router_topk(const float* __restrict__ aff, const float* __restrict__ bias,
                            int* __restrict__ lists, float* __restrict__ wts,
                            int* __restrict__ cursor, float* __restrict__ psum) {
  int n = blockIdx.x * 256 + threadIdx.x;
  if (n >= N_TOK) return;
  float a[NEXP];
  float rs = 0.f;
  for (int e = 0; e < NEXP; ++e) { a[e] = aff[n * NEXP + e]; rs += a[e]; }
  float inv = 1.f / (rs + EPS_C);
  for (int e = 0; e < NEXP; ++e) atomicAdd(&psum[e], a[e] * inv);
  int i0 = -1, i1 = -1;
  float b0 = -1e30f, b1 = -1e30f;
  for (int e = 0; e < NEXP; ++e) {
    float t = a[e] + bias[e];
    if (t > b0)      { b1 = b0; i1 = i0; b0 = t; i0 = e; }
    else if (t > b1) { b1 = t;  i1 = e; }
  }
  float g0 = a[i0], g1 = a[i1];
  float gs = 1.f / (g0 + g1 + EPS_C);
  g0 *= gs; g1 *= gs;
  int p0 = atomicAdd(&cursor[i0], 1);
  lists[i0 * N_TOK + p0] = n; wts[i0 * N_TOK + p0] = g0;
  int p1 = atomicAdd(&cursor[i1], 1);
  lists[i1 * N_TOK + p1] = n; wts[i1 * N_TOK + p1] = g1;
}

// ---------------- prefix offsets + balance loss + expert counts ----------------
__global__ void finalize_router(const int* __restrict__ cursor, const float* __restrict__ psum,
                                int* __restrict__ offs, float* __restrict__ tail) {
  if (threadIdx.x == 0 && blockIdx.x == 0) {
    int off = 0;
    float loss = 0.f;
    for (int e = 0; e < NEXP; ++e) {
      offs[e] = off; off += cursor[e];
      float f = (float)cursor[e] * (float)NEXP / (float)(TOPK * N_TOK);
      float P = psum[e] / (float)N_TOK;
      loss += f * P;
    }
    tail[0] = ALPHA_C * loss;
    int* ci = (int*)(tail + 1);
    for (int e = 0; e < NEXP; ++e) ci[e] = cursor[e];
  }
}

// ---------------- fused gate+up SwiGLU GEMM (WMMA bf16, double-buffered LDS) ----------------
__global__ __launch_bounds__(256)
void gateup_gemm(const bf16* __restrict__ X, int ldx, int Kdim,
                 const bf16* __restrict__ Bg, const bf16* __restrict__ Bu, long bStrideZ,
                 const int* __restrict__ counts, int fixedRows,
                 const int* __restrict__ rowmap, int rowcap,
                 const float* __restrict__ wts,
                 const int* __restrict__ offs,
                 bf16* __restrict__ H, int ldh) {
  __shared__ bf16 As[2][BM * LDT];
  __shared__ bf16 Bgs[2][BN * LDT];
  __shared__ bf16 Bus[2][BN * LDT];

  const int z = blockIdx.z;
  const int rows = counts ? counts[z] : fixedRows;
  const int m0 = blockIdx.y * BM;
  if (m0 >= rows) return;
  const int n0 = blockIdx.x * BN;
  const int tid = threadIdx.x;
  const int lane = tid & 31, wave = tid >> 5;
  const int wm = wave & 3, wn = wave >> 2;

  const bf16* Bgz = Bg + (long)z * bStrideZ;
  const bf16* Buz = Bu + (long)z * bStrideZ;
  const int* rmap = rowmap ? (rowmap + (long)z * rowcap) : nullptr;
  const int outBase = offs ? offs[z] : 0;

  // hoisted, loop-invariant gather addresses (the gather/predicate never runs in the K loop)
  const bf16* asrc[AITER];
  int adst[AITER];
#pragma unroll
  for (int i = 0; i < AITER; ++i) {
    int c = tid + i * 256;
    int r = c >> 3, cs = c & 7;
    adst[i] = r * LDT + cs * 8;
    int rr = m0 + r;
    asrc[i] = (rr < rows) ? (X + (long)(rmap ? rmap[rr] : rr) * ldx + cs * 8) : nullptr;
  }
  const bf16 *bgsrc[BITER], *busrc[BITER];
  int bdst[BITER];
#pragma unroll
  for (int i = 0; i < BITER; ++i) {
    int c = tid + i * 256;
    int r = c >> 3, cs = c & 7;
    bdst[i] = r * LDT + cs * 8;
    long off = (long)(n0 + r) * Kdim + cs * 8;
    bgsrc[i] = Bgz + off;
    busrc[i] = Buz + off;
  }

  auto load_tile = [&](int k0, int buf) {
#pragma unroll
    for (int i = 0; i < AITER; ++i) {
      uint4 v = {0u, 0u, 0u, 0u};
      if (asrc[i]) v = *(const uint4*)(asrc[i] + k0);
      *(uint4*)(&As[buf][adst[i]]) = v;
    }
#pragma unroll
    for (int i = 0; i < BITER; ++i) {
      *(uint4*)(&Bgs[buf][bdst[i]]) = *(const uint4*)(bgsrc[i] + k0);
      *(uint4*)(&Bus[buf][bdst[i]]) = *(const uint4*)(busrc[i] + k0);
    }
  };

  v8f accg[2][2] = {};
  v8f accu[2][2] = {};
  const int mloc = lane & 15, khalf = lane >> 4;

  load_tile(0, 0);
  __syncthreads();
  int buf = 0;
  for (int k0 = 0; k0 < Kdim; k0 += BK) {
    if (k0 + BK < Kdim) load_tile(k0 + BK, buf ^ 1);  // prefetch next tile into other buffer
#pragma unroll
    for (int kb = 0; kb < 2; ++kb) {
      v16bf afrag[2], bgfrag[2], bufrag[2];
#pragma unroll
      for (int sm = 0; sm < 2; ++sm) {
        const bf16* base = &As[buf][(wm * 32 + sm * 16 + mloc) * LDT + kb * 32 + khalf * 16];
        afrag[sm] = ld_frag(base, base + 8);
      }
#pragma unroll
      for (int sn = 0; sn < 2; ++sn) {
        const bf16* bg = &Bgs[buf][(wn * 32 + sn * 16 + mloc) * LDT + kb * 32 + khalf * 16];
        bgfrag[sn] = ld_frag(bg, bg + 8);
        const bf16* bu = &Bus[buf][(wn * 32 + sn * 16 + mloc) * LDT + kb * 32 + khalf * 16];
        bufrag[sn] = ld_frag(bu, bu + 8);
      }
#pragma unroll
      for (int sm = 0; sm < 2; ++sm)
#pragma unroll
        for (int sn = 0; sn < 2; ++sn) {
          accg[sm][sn] = __builtin_amdgcn_wmma_f32_16x16x32_bf16(
              false, afrag[sm], false, bgfrag[sn], (short)0, accg[sm][sn], false, false);
          accu[sm][sn] = __builtin_amdgcn_wmma_f32_16x16x32_bf16(
              false, afrag[sm], false, bufrag[sn], (short)0, accu[sm][sn], false, false);
        }
    }
    __syncthreads();
    buf ^= 1;
  }

  // epilogue: SwiGLU + optional per-row gate weight, store bf16
  const int nloc = lane & 15, mhalf = (lane >> 4) * 8;
#pragma unroll
  for (int sm = 0; sm < 2; ++sm)
#pragma unroll
    for (int sn = 0; sn < 2; ++sn) {
      int col = n0 + wn * 32 + sn * 16 + nloc;
#pragma unroll
      for (int j = 0; j < 8; ++j) {
        int r = m0 + wm * 32 + sm * 16 + mhalf + j;
        if (r < rows) {
          float g = accg[sm][sn][j];
          float u = accu[sm][sn][j];
          float w = wts ? wts[(long)z * rowcap + r] : 1.0f;
          float h = (g / (1.0f + __expf(-g))) * u * w;
          H[(long)(outBase + r) * ldh + col] = (bf16)h;
        }
      }
    }
}

// ---------------- down projection GEMM (WMMA bf16, double-buffered LDS) ----------------
__global__ __launch_bounds__(256)
void down_gemm(const bf16* __restrict__ Hsrc, int ldh, int Kdim,
               const bf16* __restrict__ Bd, long bStrideZ,
               const int* __restrict__ counts, int fixedRows,
               const int* __restrict__ offs,
               const int* __restrict__ rowmap, int rowcap,
               float* __restrict__ Out, int ldo) {
  __shared__ bf16 As[2][BM * LDT];
  __shared__ bf16 Bs[2][BN * LDT];

  const int z = blockIdx.z;
  const int rows = counts ? counts[z] : fixedRows;
  const int m0 = blockIdx.y * BM;
  if (m0 >= rows) return;
  const int n0 = blockIdx.x * BN;
  const int tid = threadIdx.x;
  const int lane = tid & 31, wave = tid >> 5;
  const int wm = wave & 3, wn = wave >> 2;

  const bf16* Bz = Bd + (long)z * bStrideZ;
  const int* rmap = rowmap ? (rowmap + (long)z * rowcap) : nullptr;
  const int aBase = offs ? offs[z] : 0;

  const bf16* asrc[AITER];
  int adst[AITER];
#pragma unroll
  for (int i = 0; i < AITER; ++i) {
    int c = tid + i * 256;
    int r = c >> 3, cs = c & 7;
    adst[i] = r * LDT + cs * 8;
    int rr = m0 + r;
    asrc[i] = (rr < rows) ? (Hsrc + (long)(aBase + rr) * ldh + cs * 8) : nullptr;
  }
  const bf16* bsrc[BITER];
  int bdst[BITER];
#pragma unroll
  for (int i = 0; i < BITER; ++i) {
    int c = tid + i * 256;
    int r = c >> 3, cs = c & 7;
    bdst[i] = r * LDT + cs * 8;
    bsrc[i] = Bz + (long)(n0 + r) * Kdim + cs * 8;
  }

  auto load_tile = [&](int k0, int buf) {
#pragma unroll
    for (int i = 0; i < AITER; ++i) {
      uint4 v = {0u, 0u, 0u, 0u};
      if (asrc[i]) v = *(const uint4*)(asrc[i] + k0);
      *(uint4*)(&As[buf][adst[i]]) = v;
    }
#pragma unroll
    for (int i = 0; i < BITER; ++i)
      *(uint4*)(&Bs[buf][bdst[i]]) = *(const uint4*)(bsrc[i] + k0);
  };

  v8f acc[2][2] = {};
  const int mloc = lane & 15, khalf = lane >> 4;

  load_tile(0, 0);
  __syncthreads();
  int buf = 0;
  for (int k0 = 0; k0 < Kdim; k0 += BK) {
    if (k0 + BK < Kdim) load_tile(k0 + BK, buf ^ 1);
#pragma unroll
    for (int kb = 0; kb < 2; ++kb) {
      v16bf afrag[2], bfrag[2];
#pragma unroll
      for (int sm = 0; sm < 2; ++sm) {
        const bf16* base = &As[buf][(wm * 32 + sm * 16 + mloc) * LDT + kb * 32 + khalf * 16];
        afrag[sm] = ld_frag(base, base + 8);
      }
#pragma unroll
      for (int sn = 0; sn < 2; ++sn) {
        const bf16* bb = &Bs[buf][(wn * 32 + sn * 16 + mloc) * LDT + kb * 32 + khalf * 16];
        bfrag[sn] = ld_frag(bb, bb + 8);
      }
#pragma unroll
      for (int sm = 0; sm < 2; ++sm)
#pragma unroll
        for (int sn = 0; sn < 2; ++sn)
          acc[sm][sn] = __builtin_amdgcn_wmma_f32_16x16x32_bf16(
              false, afrag[sm], false, bfrag[sn], (short)0, acc[sm][sn], false, false);
    }
    __syncthreads();
    buf ^= 1;
  }

  const int nloc = lane & 15, mhalf = (lane >> 4) * 8;
#pragma unroll
  for (int sm = 0; sm < 2; ++sm)
#pragma unroll
    for (int sn = 0; sn < 2; ++sn) {
      int col = n0 + wn * 32 + sn * 16 + nloc;
#pragma unroll
      for (int j = 0; j < 8; ++j) {
        int r = m0 + wm * 32 + sm * 16 + mhalf + j;
        if (r < rows) {
          float v = acc[sm][sn][j];
          if (rmap) atomicAdd(&Out[(long)rmap[r] * ldo + col], v);
          else      Out[(long)r * ldo + col] = v;
        }
      }
    }
}

// ---------------- host side ----------------
extern "C" void kernel_launch(void* const* d_in, const int* in_sizes, int n_in,
                              void* d_out, int out_size, void* d_ws, size_t ws_size,
                              hipStream_t stream) {
  (void)in_sizes; (void)n_in; (void)out_size; (void)ws_size;
  const float* x       = (const float*)d_in[0];
  const float* gate_w  = (const float*)d_in[1];
  const float* ebias   = (const float*)d_in[2];
  const float* sh_gate = (const float*)d_in[3];
  const float* sh_up   = (const float*)d_in[4];
  const float* sh_down = (const float*)d_in[5];
  const float* ex_gate = (const float*)d_in[6];
  const float* ex_up   = (const float*)d_in[7];
  const float* ex_down = (const float*)d_in[8];
  float* out = (float*)d_out;
  float* tail = out + (long)N_TOK * DIM;

  char* p = (char*)d_ws;
  auto alloc = [&](size_t bytes) -> char* {
    char* r = p;
    p += (bytes + 255) & ~(size_t)255;
    return r;
  };
  bf16*  xbf   = (bf16*)alloc((size_t)N_TOK * DIM * 2);
  bf16*  shTg  = (bf16*)alloc((size_t)ISH * DIM * 2);
  bf16*  shTu  = (bf16*)alloc((size_t)ISH * DIM * 2);
  bf16*  shTd  = (bf16*)alloc((size_t)DIM * ISH * 2);
  bf16*  exTg  = (bf16*)alloc((size_t)NEXP * IEXP * DIM * 2);
  bf16*  exTu  = (bf16*)alloc((size_t)NEXP * IEXP * DIM * 2);
  bf16*  exTd  = (bf16*)alloc((size_t)NEXP * DIM * IEXP * 2);
  bf16*  hsh   = (bf16*)alloc((size_t)N_TOK * ISH * 2);
  bf16*  hrt   = (bf16*)alloc((size_t)N_TOK * TOPK * IEXP * 2);
  float* aff   = (float*)alloc((size_t)N_TOK * NEXP * 4);
  int*   lists = (int*)alloc((size_t)NEXP * N_TOK * 4);
  float* wtsl  = (float*)alloc((size_t)NEXP * N_TOK * 4);
  char*  ctrl  = alloc(768);
  int*   cursor = (int*)ctrl;           // expert counts
  float* psum   = (float*)(ctrl + 256); // sum over tokens of normalized affinity
  int*   offs   = (int*)(ctrl + 512);   // exclusive prefix

  hipMemsetAsync(ctrl, 0, 768, stream);

  // preprocess: cast x; transpose+cast weights to [col][K] bf16
  cvt_f32_bf16<<<((long)N_TOK * DIM + 255) / 256, 256, 0, stream>>>(x, xbf, (long)N_TOK * DIM);
  dim3 tb(32, 8);
  transpose_cvt<<<dim3(ISH / 32, DIM / 32, 1), tb, 0, stream>>>(sh_gate, shTg, DIM, ISH);
  transpose_cvt<<<dim3(ISH / 32, DIM / 32, 1), tb, 0, stream>>>(sh_up, shTu, DIM, ISH);
  transpose_cvt<<<dim3(DIM / 32, ISH / 32, 1), tb, 0, stream>>>(sh_down, shTd, ISH, DIM);
  transpose_cvt<<<dim3(IEXP / 32, DIM / 32, NEXP), tb, 0, stream>>>(ex_gate, exTg, DIM, IEXP);
  transpose_cvt<<<dim3(IEXP / 32, DIM / 32, NEXP), tb, 0, stream>>>(ex_up, exTu, DIM, IEXP);
  transpose_cvt<<<dim3(DIM / 32, IEXP / 32, NEXP), tb, 0, stream>>>(ex_down, exTd, IEXP, DIM);

  // router
  router_aff<<<(N_TOK * NEXP + 255) / 256, 256, 0, stream>>>(x, gate_w, aff);
  router_topk<<<(N_TOK + 255) / 256, 256, 0, stream>>>(aff, ebias, lists, wtsl, cursor, psum);
  finalize_router<<<1, 32, 0, stream>>>(cursor, psum, offs, tail);

  // shared expert: h_sh = silu(x@Wg)*(x@Wu) ; out = h_sh @ Wd
  gateup_gemm<<<dim3(ISH / BN, N_TOK / BM, 1), 256, 0, stream>>>(
      xbf, DIM, DIM, shTg, shTu, 0, nullptr, N_TOK, nullptr, 0, nullptr, nullptr, hsh, ISH);
  down_gemm<<<dim3(DIM / BN, N_TOK / BM, 1), 256, 0, stream>>>(
      hsh, ISH, ISH, shTd, 0, nullptr, N_TOK, nullptr, nullptr, 0, out, DIM);

  // routed experts (token-gathered, gate weight folded into h, atomic combine)
  gateup_gemm<<<dim3(IEXP / BN, N_TOK / BM, NEXP), 256, 0, stream>>>(
      xbf, DIM, DIM, exTg, exTu, (long)IEXP * DIM, cursor, 0, lists, N_TOK, wtsl, offs, hrt, IEXP);
  down_gemm<<<dim3(DIM / BN, N_TOK / BM, NEXP), 256, 0, stream>>>(
      hrt, IEXP, IEXP, exTd, (long)DIM * IEXP, cursor, 0, offs, lists, N_TOK, out, DIM);
}
